// EfficientCrossAttention_57028575756692
// MI455X (gfx1250) — matrix operand order repaired and checked by reading, not compiled, not run on hardware
//
#include <hip/hip_runtime.h>
#include <hip/hip_bf16.h>

typedef unsigned short u16;
typedef __attribute__((ext_vector_type(16))) __bf16 v16bf;
typedef __attribute__((ext_vector_type(8)))  float  v8f;

union Frag {
  v16bf v;
  u16   h[16];
  uint4 q[2];
};

union H8 {
  uint4 q;
  u16   h[8];
};

__device__ __forceinline__ v8f wmma_bf16(v16bf a, v16bf b, v8f c) {
  // D = A(16x32 bf16) x B(32x16 bf16) + C(16x16 f32)
  return __builtin_amdgcn_wmma_f32_16x16x32_bf16(false, a, false, b, (short)0, c, false, false);
}

// Split f32 into bf16 hi + bf16 lo (error-compensated GEMM operands).
__device__ __forceinline__ void split2(float x, u16& hi, u16& lo) {
  unsigned u = __float_as_uint(x);
  hi = (u16)(u >> 16);
  float r = x - __uint_as_float(u & 0xFFFF0000u);
  lo = (u16)(__float_as_uint(r) >> 16);
}

// f32 -> bf16 round-to-nearest-even
__device__ __forceinline__ u16 f2bf(float x) {
  unsigned u = __float_as_uint(x);
  unsigned r = u + 0x7FFFu + ((u >> 16) & 1u);
  return (u16)(r >> 16);
}

// A-fragment (16x32, 16-bit) from LDS, row-major [row][K], row stride rs (halves),
// K base kb. Lane<16: M=lane, K={kb+0..15}; lane>=16: M=lane-16, K={kb+16..31}.
__device__ __forceinline__ v16bf load_afrag_lds(const u16* base, int rowBase, int rs, int kb, int lane) {
  int m  = rowBase + (lane & 15);
  int k1 = kb + ((lane >> 4) << 4);
  Frag f;
  f.q[0] = *(const uint4*)(base + m * rs + k1);
  f.q[1] = *(const uint4*)(base + m * rs + k1 + 8);
  return f.v;
}

// B-fragment (32x16, 16-bit) from global, pre-packed: [tileIdx][lane][16 halves].
__device__ __forceinline__ v16bf load_bfrag_g(const u16* g, long idxBase, int lane) {
  const u16* p = g + idxBase + (long)lane * 16;
  Frag f;
  f.q[0] = *(const uint4*)(p);
  f.q[1] = *(const uint4*)(p + 8);
  return f.v;
}

// ---------------------------------------------------------------------------
// Pack W (rows = Nout, cols = K) scaled by layernorm gamma into B-fragment
// bf16 hi/lo layout: out[((nt*KS + ks)*32 + lane)*16 + e]
//   N = nt*16 + (lane&15),  K = ks*32 + (lane>=16 ? 16 : 0) + e
// ---------------------------------------------------------------------------
__global__ __launch_bounds__(256) void k_pack(const float* __restrict__ W,
                                              const float* __restrict__ g,
                                              u16* __restrict__ ph, u16* __restrict__ pl,
                                              int K, int NT) {
  int tid = blockIdx.x * 256 + threadIdx.x;
  int KS = K >> 5;
  int total = NT * KS * 512;
  if (tid >= total) return;
  int e    = tid & 15;
  int lane = (tid >> 4) & 31;
  int rest = tid >> 9;           // = nt*KS + ks
  int ks = rest % KS;
  int nt = rest / KS;
  int n  = nt * 16 + (lane & 15);
  int kk = ks * 32 + ((lane >> 4) << 4) + e;
  float v = W[(long)n * K + kk] * g[kk];
  u16 hi, lo;
  split2(v, hi, lo);
  ph[tid] = hi;
  pl[tid] = lo;
}

// u[n] = sum_k W[n,k]*g[k];  c[n] = bias[n] + sum_k W[n,k]*beta[k]
__global__ __launch_bounds__(256) void k_uc(const float* __restrict__ W,
                                            const float* __restrict__ g,
                                            const float* __restrict__ bet,
                                            const float* __restrict__ bias,
                                            float* __restrict__ u, float* __restrict__ c,
                                            int K, int Nout) {
  int n = blockIdx.x * 256 + threadIdx.x;
  if (n >= Nout) return;
  float su = 0.f, sc = 0.f;
  for (int k = 0; k < K; ++k) {
    float wv = W[(long)n * K + k];
    su += wv * g[k];
    sc += wv * bet[k];
  }
  u[n] = su;
  c[n] = sc + bias[n];
}

// ---------------------------------------------------------------------------
// Fused LayerNorm + projection via fixup:  out = rstd*( (W*g) x - mean*u ) + c
// 64 rows x 512 cols per block, 8 waves (wave = 16 rows x 256 cols = 16 tiles).
// Error-compensated bf16: 3 WMMAs per tile per K-slice.
// ---------------------------------------------------------------------------
__global__ __launch_bounds__(256) void k_proj(const float* __restrict__ src,
                                              const u16* __restrict__ wh, const u16* __restrict__ wl,
                                              const float* __restrict__ u, const float* __restrict__ c,
                                              float* __restrict__ outp, int nRows, int K) {
  extern __shared__ char smem[];
  float* stats = (float*)smem;               // 64 x {sum,sumsq} -> {mean,rstd}
  u16* sxh = (u16*)(smem + 512);             // [64][40] staged hi
  u16* sxl = sxh + 64 * 40;                  // [64][40] staged lo
  const int KS = K >> 5;
  const float invK = 1.0f / (float)K;
  const int tid = threadIdx.x, lane = tid & 31, w = tid >> 5;
  const int Mg = w >> 1, Ng = w & 1;
  const int hi8 = (lane >> 4) << 3;
  const long rowBlock = (long)blockIdx.x * 64;
  const int srow = tid >> 2;
  const int scol = (tid & 3) << 3;
  const bool valid = (rowBlock + srow) < nRows;

  if (tid < 128) stats[tid] = 0.f;
  v8f zerov = {};
  v8f acc[16];
#pragma unroll
  for (int i = 0; i < 16; ++i) acc[i] = zerov;

  float s1 = 0.f, s2 = 0.f;
  for (int ks = 0; ks < KS; ++ks) {
    __syncthreads();
    float vals[8];
#pragma unroll
    for (int i = 0; i < 8; ++i) vals[i] = 0.f;
    if (valid) {
      const float* p = src + (rowBlock + srow) * (long)K + ks * 32 + scol;
      float4 a0 = *(const float4*)(p);
      float4 a1 = *(const float4*)(p + 4);
      vals[0] = a0.x; vals[1] = a0.y; vals[2] = a0.z; vals[3] = a0.w;
      vals[4] = a1.x; vals[5] = a1.y; vals[6] = a1.z; vals[7] = a1.w;
    }
    H8 fh, fl;
#pragma unroll
    for (int i = 0; i < 8; ++i) {
      float v = vals[i];
      s1 += v; s2 += v * v;
      split2(v, fh.h[i], fl.h[i]);
    }
    *(uint4*)(sxh + srow * 40 + scol) = fh.q;
    *(uint4*)(sxl + srow * 40 + scol) = fl.q;
    __syncthreads();

    v16bf ah = load_afrag_lds(sxh, Mg * 16, 40, 0, lane);
    v16bf al = load_afrag_lds(sxl, Mg * 16, 40, 0, lane);
#pragma unroll
    for (int nt = 0; nt < 16; ++nt) {
      long bbase = (long)((Ng * 16 + nt) * KS + ks) * 512;
      v16bf bh = load_bfrag_g(wh, bbase, lane);
      v16bf bl = load_bfrag_g(wl, bbase, lane);
      acc[nt] = wmma_bf16(ah, bh, acc[nt]);
      acc[nt] = wmma_bf16(ah, bl, acc[nt]);
      acc[nt] = wmma_bf16(al, bh, acc[nt]);
    }
  }

  __syncthreads();
  atomicAdd(&stats[srow * 2 + 0], s1);
  atomicAdd(&stats[srow * 2 + 1], s2);
  __syncthreads();
  if (tid < 64) {
    float m = stats[tid * 2] * invK;
    float var = stats[tid * 2 + 1] * invK - m * m;
    stats[tid * 2 + 0] = m;
    stats[tid * 2 + 1] = rsqrtf(var + 1e-5f);
  }
  __syncthreads();

  float mean_r[8], rstd_r[8];
#pragma unroll
  for (int r = 0; r < 8; ++r) {
    int rl = Mg * 16 + r + hi8;
    mean_r[r] = stats[rl * 2 + 0];
    rstd_r[r] = stats[rl * 2 + 1];
  }
#pragma unroll
  for (int nt = 0; nt < 16; ++nt) {
    int col = Ng * 256 + nt * 16 + (lane & 15);
    float uv = u[col], cv = c[col];
#pragma unroll
    for (int r = 0; r < 8; ++r) {
      float q = rstd_r[r] * (acc[nt][r] - mean_r[r] * uv) + cv;
      long grow = rowBlock + Mg * 16 + r + hi8;
      if (grow < nRows) outp[grow * 512 + col] = q;
    }
  }
}

// ---------------------------------------------------------------------------
// Per (b,h): softmax over N=77 on k columns, att[d,l] = sum_n k_sm[n,d]*v[n,l],
// packed directly into B-fragment bf16 layout for the second GEMM.
// ---------------------------------------------------------------------------
__global__ __launch_bounds__(256) void k_att(const float* __restrict__ kraw,
                                             const float* __restrict__ vraw,
                                             u16* __restrict__ attp) {
  __shared__ float kc[77 * 64];
  __shared__ float vc[77 * 64];
  const int bh = blockIdx.x;
  const int b = bh >> 3, h = bh & 7;
  const int tid = threadIdx.x;
  for (int idx = tid; idx < 77 * 64; idx += 256) {
    int n = idx >> 6, j = idx & 63;
    long src = (long)(b * 77 + n) * 512 + h * 64 + j;
    kc[idx] = kraw[src];
    vc[idx] = vraw[src];
  }
  __syncthreads();
  if (tid < 64) {
    float mx = -3.0e38f;
    for (int n = 0; n < 77; ++n) mx = fmaxf(mx, kc[n * 64 + tid]);
    float s = 0.f;
    for (int n = 0; n < 77; ++n) {
      float e = __expf(kc[n * 64 + tid] - mx);
      kc[n * 64 + tid] = e;
      s += e;
    }
    float inv = 1.0f / s;
    for (int n = 0; n < 77; ++n) kc[n * 64 + tid] *= inv;
  }
  __syncthreads();
#pragma unroll
  for (int i = 0; i < 16; ++i) {
    int lin = tid + 256 * i;
    int d = lin >> 6, l = lin & 63;
    float s = 0.f;
    for (int n = 0; n < 77; ++n) s += kc[n * 64 + d] * vc[n * 64 + l];
    // pack into B-fragment: slice = d/32, lane = ((d>>4)&1)*16 + (l&15), e = d&15, tile = l/16
    int sl = d >> 5;
    int e  = d & 15;
    int lane = (((d >> 4) & 1) << 4) + (l & 15);
    int tl = l >> 4;
    long o = ((((long)bh * 2 + sl) * 4 + tl) * 512) + lane * 16 + e;
    attp[o] = f2bf(s);
  }
}

// ---------------------------------------------------------------------------
// Main fused kernel: LN(x) -> Q projection (WMMA, bf16 hi/lo) -> +bias via c ->
// per-head softmax over 64 -> q @ att (WMMA) -> out = x + y.
// 64 rows x 512 cols per block; K = 512 (16 slices).
// ---------------------------------------------------------------------------
__global__ __launch_bounds__(256) void k_main(const float* __restrict__ x,
                                              const u16* __restrict__ wh, const u16* __restrict__ wl,
                                              const float* __restrict__ u, const float* __restrict__ c,
                                              const u16* __restrict__ attp,
                                              float* __restrict__ outp) {
  extern __shared__ char smem[];
  float* stats = (float*)smem;               // 64 x {sum,sumsq} -> {mean,rstd}
  u16* sxh = (u16*)(smem + 512);             // [64][40]
  u16* sxl = sxh + 64 * 40;                  // [64][40]
  u16* sq  = sxl + 64 * 40;                  // 8 waves x [16][272]
  const int KS = 16;
  const int tid = threadIdx.x, lane = tid & 31, w = tid >> 5;
  const int Mg = w >> 1, Ng = w & 1;
  const int hi8 = (lane >> 4) << 3;
  const long rowBlock = (long)blockIdx.x * 64;
  const int b = blockIdx.x >> 6;             // T=4096 => 64 blocks per batch
  const int srow = tid >> 2;
  const int scol = (tid & 3) << 3;

  if (tid < 128) stats[tid] = 0.f;
  v8f zerov = {};
  v8f acc[16];
#pragma unroll
  for (int i = 0; i < 16; ++i) acc[i] = zerov;

  float s1 = 0.f, s2 = 0.f;
  for (int ks = 0; ks < KS; ++ks) {
    __syncthreads();
    const float* p = x + (rowBlock + srow) * 512 + ks * 32 + scol;
    float4 a0 = *(const float4*)(p);
    float4 a1 = *(const float4*)(p + 4);
    float vals[8] = {a0.x, a0.y, a0.z, a0.w, a1.x, a1.y, a1.z, a1.w};
    H8 fh, fl;
#pragma unroll
    for (int i = 0; i < 8; ++i) {
      float v = vals[i];
      s1 += v; s2 += v * v;
      split2(v, fh.h[i], fl.h[i]);
    }
    *(uint4*)(sxh + srow * 40 + scol) = fh.q;
    *(uint4*)(sxl + srow * 40 + scol) = fl.q;
    __syncthreads();

    v16bf ah = load_afrag_lds(sxh, Mg * 16, 40, 0, lane);
    v16bf al = load_afrag_lds(sxl, Mg * 16, 40, 0, lane);
#pragma unroll
    for (int nt = 0; nt < 16; ++nt) {
      long bbase = (long)((Ng * 16 + nt) * KS + ks) * 512;
      v16bf bh = load_bfrag_g(wh, bbase, lane);
      v16bf bl = load_bfrag_g(wl, bbase, lane);
      acc[nt] = wmma_bf16(ah, bh, acc[nt]);
      acc[nt] = wmma_bf16(ah, bl, acc[nt]);
      acc[nt] = wmma_bf16(al, bh, acc[nt]);
    }
  }

  // finalize row stats
  __syncthreads();
  atomicAdd(&stats[srow * 2 + 0], s1);
  atomicAdd(&stats[srow * 2 + 1], s2);
  __syncthreads();
  if (tid < 64) {
    float m = stats[tid * 2] * (1.0f / 512.0f);
    float var = stats[tid * 2 + 1] * (1.0f / 512.0f) - m * m;
    stats[tid * 2 + 0] = m;
    stats[tid * 2 + 1] = rsqrtf(var + 1e-5f);
  }
  __syncthreads();

  // fixup: q = rstd*(acc - mean*u) + c   (c includes q_b and beta terms)
  float mean_r[8], rstd_r[8];
#pragma unroll
  for (int r = 0; r < 8; ++r) {
    int rl = Mg * 16 + r + hi8;
    mean_r[r] = stats[rl * 2 + 0];
    rstd_r[r] = stats[rl * 2 + 1];
  }
#pragma unroll
  for (int nt = 0; nt < 16; ++nt) {
    int col = Ng * 256 + nt * 16 + (lane & 15);
    float uv = u[col], cv = c[col];
#pragma unroll
    for (int r = 0; r < 8; ++r)
      acc[nt][r] = rstd_r[r] * (acc[nt][r] - mean_r[r] * uv) + cv;
  }

  // per-head softmax over 64 columns (4 tiles x 16 lanes within half-wave)
#pragma unroll
  for (int h = 0; h < 4; ++h) {
#pragma unroll
    for (int r = 0; r < 8; ++r) {
      float mx = acc[h * 4 + 0][r];
#pragma unroll
      for (int t = 1; t < 4; ++t) mx = fmaxf(mx, acc[h * 4 + t][r]);
#pragma unroll
      for (int msk = 1; msk <= 8; msk <<= 1) mx = fmaxf(mx, __shfl_xor(mx, msk, 32));
      float s = 0.f;
#pragma unroll
      for (int t = 0; t < 4; ++t) {
        float e = __expf(acc[h * 4 + t][r] - mx);
        acc[h * 4 + t][r] = e;
        s += e;
      }
#pragma unroll
      for (int msk = 1; msk <= 8; msk <<= 1) s += __shfl_xor(s, msk, 32);
      float inv = 1.0f / s;
#pragma unroll
      for (int t = 0; t < 4; ++t) acc[h * 4 + t][r] *= inv;
    }
  }

  // stage softmaxed q (wave-private LDS region) for A-fragment reload
  u16* sqw = sq + w * (16 * 272);
#pragma unroll
  for (int nt = 0; nt < 16; ++nt) {
#pragma unroll
    for (int r = 0; r < 8; ++r)
      sqw[(r + hi8) * 272 + nt * 16 + (lane & 15)] = f2bf(acc[nt][r]);
  }
  __syncthreads();

  // second GEMM: y[rows, head l] = q_head(16x64) @ att_head(64x64), + residual
#pragma unroll
  for (int h = 0; h < 4; ++h) {
    v16bf aq0 = load_afrag_lds(sqw, 0, 272, h * 64, lane);
    v16bf aq1 = load_afrag_lds(sqw, 0, 272, h * 64 + 32, lane);
    long bhbase = (long)(b * 8 + Ng * 4 + h) * 8 * 512;   // (bh*2 slices)*4 tiles*512
#pragma unroll
    for (int t = 0; t < 4; ++t) {
      v16bf b0 = load_bfrag_g(attp, bhbase + (long)t * 512, lane);
      v16bf b1 = load_bfrag_g(attp, bhbase + (long)(4 + t) * 512, lane);
      v8f y = zerov;
      y = wmma_bf16(aq0, b0, y);
      y = wmma_bf16(aq1, b1, y);
      int col = Ng * 256 + h * 64 + t * 16 + (lane & 15);
#pragma unroll
      for (int r = 0; r < 8; ++r) {
        long off = (rowBlock + Mg * 16 + r + hi8) * 512 + col;
        outp[off] = x[off] + y[r];
      }
    }
  }
}

// ---------------------------------------------------------------------------

extern "C" void kernel_launch(void* const* d_in, const int* in_sizes, int n_in,
                              void* d_out, int out_size, void* d_ws, size_t ws_size,
                              hipStream_t stream) {
  const float* x   = (const float*)d_in[0];
  const float* xf  = (const float*)d_in[1];
  const float* q_w = (const float*)d_in[2];
  const float* q_b = (const float*)d_in[3];
  const float* k_w = (const float*)d_in[4];
  const float* k_b = (const float*)d_in[5];
  const float* v_w = (const float*)d_in[6];
  const float* v_b = (const float*)d_in[7];
  const float* ng  = (const float*)d_in[8];
  const float* nb  = (const float*)d_in[9];
  const float* tg  = (const float*)d_in[10];
  const float* tb  = (const float*)d_in[11];
  float* out = (float*)d_out;
  (void)in_sizes; (void)n_in; (void)out_size; (void)ws_size;

  // workspace layout
  u16* qwh = (u16*)d_ws;                       // 512*512
  u16* qwl = qwh + 512 * 512;
  u16* kwh = qwl + 512 * 512;                  // 512*768
  u16* kwl = kwh + 512 * 768;
  u16* vwh = kwl + 512 * 768;
  u16* vwl = vwh + 512 * 768;
  float* qu = (float*)(vwl + 512 * 768);
  float* qc = qu + 512;
  float* ku = qc + 512;
  float* kc = ku + 512;
  float* vu = kc + 512;
  float* vc = vu + 512;
  float* kraw = vc + 512;                      // [1232][512]
  float* vraw = kraw + 1232 * 512;
  u16* attp = (u16*)(vraw + 1232 * 512);       // B*H * 2*4*32*16 halves

  // 1) pack layernorm-scaled weights into bf16 hi/lo B-fragment layout
  k_pack<<<(32 * 16 * 512 + 255) / 256, 256, 0, stream>>>(q_w, ng, qwh, qwl, 512, 32);
  k_pack<<<(32 * 24 * 512 + 255) / 256, 256, 0, stream>>>(k_w, tg, kwh, kwl, 768, 32);
  k_pack<<<(32 * 24 * 512 + 255) / 256, 256, 0, stream>>>(v_w, tg, vwh, vwl, 768, 32);

  // 2) layernorm fixup vectors (fold beta + bias)
  k_uc<<<2, 256, 0, stream>>>(q_w, ng, nb, q_b, qu, qc, 512, 512);
  k_uc<<<2, 256, 0, stream>>>(k_w, tg, tb, k_b, ku, kc, 768, 512);
  k_uc<<<2, 256, 0, stream>>>(v_w, tg, tb, v_b, vu, vc, 768, 512);

  // 3) fused LN + K/V projections (rows = B*N = 1232, K = 768)
  size_t lds_proj = 512 + 2 * 64 * 40 * 2;     // stats + hi/lo staging
  k_proj<<<20, 256, lds_proj, stream>>>(xf, kwh, kwl, ku, kc, kraw, 1232, 768);
  k_proj<<<20, 256, lds_proj, stream>>>(xf, vwh, vwl, vu, vc, vraw, 1232, 768);

  // 4) softmax over N + att, packed for second GEMM
  k_att<<<16 * 8, 256, 0, stream>>>(kraw, vraw, attp);

  // 5) main fused kernel over B*T = 65536 rows
  size_t lds_main = 512 + 2 * 64 * 40 * 2 + 8 * 16 * 272 * 2;
  k_main<<<1024, 256, lds_main, stream>>>(x, qwh, qwl, qu, qc, attp, out);
}
